// MultiheadAttention_21758304321961
// MI455X (gfx1250) — matrix operand order repaired and checked by reading, not compiled, not run on hardware
//
#include <hip/hip_runtime.h>
#include <hip/hip_bf16.h>
#include <math.h>
#include <stdint.h>

typedef __bf16 bf16_t;
typedef __attribute__((ext_vector_type(16))) __bf16 v16bf;
typedef __attribute__((ext_vector_type(4)))  __bf16 bf16x4;
typedef __attribute__((ext_vector_type(8)))  float   v8f;
typedef __attribute__((ext_vector_type(4)))  float   f32x4;

union FragBF {
    v16bf v;
    f32x4 q[2];
};

// ---------------------------------------------------------------------------
// WMMA helper: D = A(16x32 bf16) * B(32x16 bf16) + C(16x16 f32)
// ---------------------------------------------------------------------------
__device__ __forceinline__ v8f wmma_bf16(v16bf a, v16bf b, v8f c) {
    return __builtin_amdgcn_wmma_f32_16x16x32_bf16(
        /*neg_a=*/false, a, /*neg_b=*/false, b,
        /*c_mod=*/(short)0, c, /*reuse_a=*/false, /*reuse_b=*/false);
}

// ---------------------------------------------------------------------------
// A fragment (16x32, bf16, row-major source, leading dim `ld` elements).
// Layout: lane<16 holds row m=lane, K = 0..15;
//         lane>=16 holds row m=lane-16, K = 16..31.
// ---------------------------------------------------------------------------
__device__ __forceinline__ v16bf load_frag_a(const bf16_t* base, int ld) {
    const int lane = threadIdx.x & 31;
    const int m    = lane & 15;
    const int kb   = (lane >> 4) << 4;           // 0 or 16
    const bf16_t* p = base + (size_t)m * ld + kb;
    FragBF f;
    f.q[0] = *reinterpret_cast<const f32x4*>(p);       // K = kb .. kb+7
    f.q[1] = *reinterpret_cast<const f32x4*>(p + 8);   // K = kb+8 .. kb+15
    return f.v;
}

// ---------------------------------------------------------------------------
// B fragment (32x16 bf16) with B[k][n] = Wt[n][k], Wt row-major, ldim `ld`.
// Layout (contiguous k, same as A): lane<16 holds col n=lane, K=0..15; lane>=16 col n, K=16..31.
// Per-lane data is 16 contiguous bf16 -> two b128 loads.
// ---------------------------------------------------------------------------
__device__ __forceinline__ v16bf load_frag_bt(const bf16_t* base, int ld) {
    const int lane = threadIdx.x & 31;
    const int n    = lane & 15;
    const int kb   = (lane >> 4) << 4;           // 0 or 16
    const bf16_t* p = base + (size_t)n * ld + kb;
    FragBF f;
    f.q[0] = *reinterpret_cast<const f32x4*>(p);
    f.q[1] = *reinterpret_cast<const f32x4*>(p + 8);
    return f.v;
}

// ---------------------------------------------------------------------------
// CDNA5 async global->LDS copy (ASYNCcnt). vdst VGPR = LDS byte offset
// (flat addr[31:0] is the wave-relative LDS offset; aperture is [63:32]).
// ---------------------------------------------------------------------------
__device__ __forceinline__ void async_ld_b128(uint32_t lds_off, const void* g) {
    asm volatile("global_load_async_to_lds_b128 %0, %1, off"
                 :: "v"(lds_off), "v"(g) : "memory");
}
__device__ __forceinline__ void wait_asynccnt0() {
    asm volatile("s_wait_asynccnt 0x0" ::: "memory");
}

// Stage a 128-row x 64-col bf16 tile (global row stride ldg) into LDS
// (row stride 64). 256 threads, 4 async b128 copies each.
__device__ __forceinline__ void async_stage_128x64(bf16_t* lds_base,
                                                   const bf16_t* g_base,
                                                   int ldg) {
    const uint32_t base = (uint32_t)(uintptr_t)lds_base;
#pragma unroll
    for (int i = 0; i < 4; ++i) {
        const int c   = (int)threadIdx.x + i * 256;  // 1024 chunks of 8 bf16
        const int row = c >> 3;
        const int col = (c & 7) << 3;
        async_ld_b128(base + (uint32_t)(row * 64 + col) * 2u,
                      g_base + (size_t)row * ldg + col);
    }
}

// Stage a 64-row x 64-col bf16 tile into LDS (row stride 64); 256 threads,
// 2 async b128 copies each.
__device__ __forceinline__ void async_stage_64x64(bf16_t* lds_base,
                                                  const bf16_t* g_base,
                                                  int ldg) {
    const uint32_t base = (uint32_t)(uintptr_t)lds_base;
#pragma unroll
    for (int i = 0; i < 2; ++i) {
        const int c   = (int)threadIdx.x + i * 256;  // 512 chunks of 8 bf16
        const int row = c >> 3;
        const int col = (c & 7) << 3;
        async_ld_b128(base + (uint32_t)(row * 64 + col) * 2u,
                      g_base + (size_t)row * ldg + col);
    }
}

// ---------------------------------------------------------------------------
// fp32 -> bf16 conversion, 4-wide
// ---------------------------------------------------------------------------
__global__ void cvt_f32_bf16_kernel(const float* __restrict__ in,
                                    bf16_t* __restrict__ out, int n4) {
    int i = blockIdx.x * blockDim.x + threadIdx.x;
    if (i < n4) {
        f32x4 v = reinterpret_cast<const f32x4*>(in)[i];
        bf16x4 o;
#pragma unroll
        for (int j = 0; j < 4; ++j) o[j] = (bf16_t)v[j];
        reinterpret_cast<bf16x4*>(out)[i] = o;
    }
}

// ---------------------------------------------------------------------------
// C[M,N] = A[M,K](bf16) * W[N,K]^T (bf16), fp32 accumulate.
// Block: 256 threads = 8 waves; block tile 128x128; wave tile 32x64.
// LDS double-buffered, staged K-depth 64, async global->LDS prefetch.
// ---------------------------------------------------------------------------
template <bool OUT_BF16>
__global__ __launch_bounds__(256)
void gemm_xwt_kernel(const bf16_t* __restrict__ A,
                     const bf16_t* __restrict__ W,
                     void* __restrict__ Cout,
                     int M, int N, int K) {
    const int wid  = threadIdx.x >> 5;
    const int lane = threadIdx.x & 31;
    const int wm   = wid & 3;            // 0..3 -> 32-row strips
    const int wn   = wid >> 2;           // 0..1 -> 64-col strips
    const int mb   = blockIdx.y * 128;
    const int nb   = blockIdx.x * 128;

    __shared__ __align__(16) bf16_t Abuf[2][128][64];  // 32 KiB
    __shared__ __align__(16) bf16_t Bbuf[2][128][64];  // 32 KiB

    v8f acc[2][4] = {};

    const int nStages = K / 64;

    // Prologue: stage k-chunk 0 into buffer 0
    async_stage_128x64(&Abuf[0][0][0], A + (size_t)mb * K, K);
    async_stage_128x64(&Bbuf[0][0][0], W + (size_t)nb * K, K);
    wait_asynccnt0();
    __syncthreads();

    for (int st = 0; st < nStages; ++st) {
        const int cur = st & 1;
        // Prefetch next stage while computing on current (overlap with WMMAs)
        if (st + 1 < nStages) {
            const int k0 = (st + 1) * 64;
            async_stage_128x64(&Abuf[cur ^ 1][0][0], A + (size_t)mb * K + k0, K);
            async_stage_128x64(&Bbuf[cur ^ 1][0][0], W + (size_t)nb * K + k0, K);
        }

        // 2 k-steps of 32 from the staged 64-deep chunk -> 16 WMMAs/wave
#pragma unroll
        for (int ks = 0; ks < 2; ++ks) {
            v16bf a0 = load_frag_a(&Abuf[cur][wm * 32][ks * 32], 64);
            v16bf a1 = load_frag_a(&Abuf[cur][wm * 32 + 16][ks * 32], 64);
#pragma unroll
            for (int nt = 0; nt < 4; ++nt) {
                v16bf bb = load_frag_bt(&Bbuf[cur][wn * 64 + nt * 16][ks * 32], 64);
                acc[0][nt] = wmma_bf16(a0, bb, acc[0][nt]);
                acc[1][nt] = wmma_bf16(a1, bb, acc[1][nt]);
            }
        }

        if (st + 1 < nStages) wait_asynccnt0();
        __syncthreads();
    }

    const int hi = lane >> 4;
    const int nc = lane & 15;
    const int m0 = mb + wm * 32;
    const int n0 = nb + wn * 64;
#pragma unroll
    for (int part = 0; part < 2; ++part) {
#pragma unroll
        for (int nt = 0; nt < 4; ++nt) {
#pragma unroll
            for (int r = 0; r < 8; ++r) {
                const int gm = m0 + part * 16 + r + 8 * hi;
                const int gn = n0 + nt * 16 + nc;
                const size_t idx = (size_t)gm * N + gn;
                if (OUT_BF16)
                    ((bf16_t*)Cout)[idx] = (bf16_t)acc[part][nt][r];
                else
                    ((float*)Cout)[idx] = acc[part][nt][r];
            }
        }
    }
}

// ---------------------------------------------------------------------------
// Flash attention (causal). qkv: [B,S,3D] bf16 (Q at +0, K at +1024, V at
// +2048 within each row; head h occupies cols h*64..h*64+63 of each third).
// O: [B,S,D] bf16 context.
// Block = 256 threads (8 waves); one block per (b, h, 128-query tile).
// Wave w owns query rows q0+16w .. q0+16w+15. K/V tiles of 64 keys are
// async-prefetched into LDS one iteration ahead (double-buffered), so the
// steady-state compute loop touches only LDS.
// ---------------------------------------------------------------------------
__global__ __launch_bounds__(256)
void attn_kernel(const bf16_t* __restrict__ qkv, bf16_t* __restrict__ O) {
    constexpr int S  = 2048;
    constexpr int D3 = 3072;
    constexpr int Dm = 1024;

    const int qtile = blockIdx.x;           // 0..15 (128-query tiles)
    const int h     = blockIdx.y;           // 0..15
    const int b     = blockIdx.z;           // 0..3
    const int wid   = threadIdx.x >> 5;     // 0..7
    const int lane  = threadIdx.x & 31;
    const int hi    = lane >> 4;
    const int nc    = lane & 15;
    const int q0    = qtile * 128;

    const bf16_t* Qb = qkv + (size_t)b * S * D3 + h * 64;
    const bf16_t* Kb = Qb + 1024;
    const bf16_t* Vb = Qb + 2048;

    __shared__ __align__(16) bf16_t Kbuf[2][64][64];   // 16 KiB (async staged)
    __shared__ __align__(16) bf16_t Vrow[2][64][64];   // 16 KiB (async staged)
    __shared__ __align__(16) bf16_t Vt[64][72];        // 9 KiB  V^T (padded)
    __shared__ __align__(16) bf16_t Pbuf[8][16][64];   // 16 KiB per-wave P

    // Preload Q fragments (reused across whole K loop)
    v16bf qa[2];
    {
        const bf16_t* qrow = Qb + (size_t)(q0 + wid * 16) * D3;
        qa[0] = load_frag_a(qrow, D3);        // d = 0..31
        qa[1] = load_frag_a(qrow + 32, D3);   // d = 32..63
    }

    v8f   o[4] = {};
    float mst[8], lst[8];
#pragma unroll
    for (int r = 0; r < 8; ++r) { mst[r] = -1e30f; lst[r] = 0.0f; }

    const float scale = 0.125f;  // 1/sqrt(64)
    const int kt_end = 2 * qtile + 1;       // causal: keys up to q0+127

    // Prologue: async-stage K/V tile 0 into buffer 0
    async_stage_64x64(&Kbuf[0][0][0], Kb, D3);
    async_stage_64x64(&Vrow[0][0][0], Vb, D3);
    wait_asynccnt0();

    for (int kt = 0; kt <= kt_end; ++kt) {
        const int cur = kt & 1;
        const int kb0 = kt * 64;

        __syncthreads();  // staged buffers visible; prior Vt reads complete

        // LDS->LDS transpose of the staged V tile: Vt[d][key] = Vrow[key][d]
        for (int i = threadIdx.x; i < 512; i += 256) {
            const int key = i >> 3;
            const int d8  = (i & 7) << 3;
            union { f32x4 v; bf16_t e[8]; } u;
            u.v = *reinterpret_cast<const f32x4*>(&Vrow[cur][key][d8]);
#pragma unroll
            for (int j = 0; j < 8; ++j) Vt[d8 + j][key] = u.e[j];
        }

        // Prefetch next K/V tile into the other buffer (overlaps compute)
        if (kt + 1 <= kt_end) {
            const size_t nk = (size_t)(kt + 1) * 64;
            async_stage_64x64(&Kbuf[cur ^ 1][0][0], Kb + nk * D3, D3);
            async_stage_64x64(&Vrow[cur ^ 1][0][0], Vb + nk * D3, D3);
        }

        __syncthreads();  // Vt ready

        // ---- scores S = Q K^T (16x64 per wave), fragments from LDS ----
        v8f s[4];
#pragma unroll
        for (int nt = 0; nt < 4; ++nt) {
            v16bf b0 = load_frag_bt(&Kbuf[cur][nt * 16][0], 64);   // d 0..31
            v16bf b1 = load_frag_bt(&Kbuf[cur][nt * 16][32], 64);  // d 32..63
            v8f z = {};
            z = wmma_bf16(qa[0], b0, z);
            z = wmma_bf16(qa[1], b1, z);
            s[nt] = z;
        }

        // ---- mask + online softmax ----
        float p[4][8];
#pragma unroll
        for (int r = 0; r < 8; ++r) {
            const int qg = q0 + wid * 16 + r + 8 * hi;
            float rm = -1e30f;
#pragma unroll
            for (int nt = 0; nt < 4; ++nt) {
                const int kg = kb0 + nt * 16 + nc;
                float v = s[nt][r] * scale;
                if (kg > qg) v = -1e30f;   // causal mask
                p[nt][r] = v;
                rm = fmaxf(rm, v);
            }
            // row max across the 16 lanes of this half-wave
#pragma unroll
            for (int off = 1; off < 16; off <<= 1)
                rm = fmaxf(rm, __shfl_xor(rm, off, 32));
            const float mnew = fmaxf(mst[r], rm);

            float rs = 0.0f;
#pragma unroll
            for (int nt = 0; nt < 4; ++nt) {
                const float e = __expf(p[nt][r] - mnew);
                p[nt][r] = e;
                Pbuf[wid][r + 8 * hi][nt * 16 + nc] = (bf16_t)e;
                rs += e;
            }
#pragma unroll
            for (int off = 1; off < 16; off <<= 1)
                rs += __shfl_xor(rs, off, 32);

            const float alpha = __expf(mst[r] - mnew);
            lst[r] = lst[r] * alpha + rs;
            mst[r] = mnew;
#pragma unroll
            for (int nt = 0; nt < 4; ++nt)
                o[nt][r] *= alpha;
        }

        // ---- O += P * V (fragments from LDS) ----
#pragma unroll
        for (int ks = 0; ks < 2; ++ks) {
            v16bf pa = load_frag_a(&Pbuf[wid][0][ks * 32], 64);
#pragma unroll
            for (int nt = 0; nt < 4; ++nt) {
                v16bf vb = load_frag_bt(&Vt[nt * 16][ks * 32], 72);
                o[nt] = wmma_bf16(pa, vb, o[nt]);
            }
        }

        wait_asynccnt0();  // next tile's async copies complete (own wave)
    }

    // ---- normalize + store context [B,S,D] ----
#pragma unroll
    for (int r = 0; r < 8; ++r) {
        const float inv = 1.0f / lst[r];
        const int qg = q0 + wid * 16 + r + 8 * hi;
        const size_t row = ((size_t)b * S + qg) * Dm + h * 64;
#pragma unroll
        for (int nt = 0; nt < 4; ++nt)
            O[row + nt * 16 + nc] = (bf16_t)(o[nt][r] * inv);
    }
}

// ---------------------------------------------------------------------------
// Host launcher
// ---------------------------------------------------------------------------
extern "C" void kernel_launch(void* const* d_in, const int* in_sizes, int n_in,
                              void* d_out, int out_size, void* d_ws, size_t ws_size,
                              hipStream_t stream) {
    (void)in_sizes; (void)n_in; (void)out_size; (void)ws_size;

    constexpr int B = 4, S = 2048, D = 1024;
    constexpr int M  = B * S;      // 8192
    constexpr int N3 = 3 * D;      // 3072

    const float* x     = (const float*)d_in[0];   // [B,S,D]
    const float* w_qkv = (const float*)d_in[1];   // [3D,D]
    const float* w_out = (const float*)d_in[2];   // [D,D]
    float* out = (float*)d_out;                   // [B,S,D]

    const size_t nX   = (size_t)M * D;     // 8388608
    const size_t nWq  = (size_t)N3 * D;    // 3145728
    const size_t nWo  = (size_t)D * D;     // 1048576
    const size_t nQKV = (size_t)M * N3;    // 25165824

    bf16_t* xb    = (bf16_t*)d_ws;
    bf16_t* wqkvb = xb + nX;
    bf16_t* woutb = wqkvb + nWq;
    bf16_t* qkvb  = woutb + nWo;
    bf16_t* ob    = qkvb + nQKV;           // total ~88 MiB of d_ws

    // 1) fp32 -> bf16 conversions (4-wide)
    cvt_f32_bf16_kernel<<<(int)((nX / 4 + 255) / 256), 256, 0, stream>>>(x, xb, (int)(nX / 4));
    cvt_f32_bf16_kernel<<<(int)((nWq / 4 + 255) / 256), 256, 0, stream>>>(w_qkv, wqkvb, (int)(nWq / 4));
    cvt_f32_bf16_kernel<<<(int)((nWo / 4 + 255) / 256), 256, 0, stream>>>(w_out, woutb, (int)(nWo / 4));

    // 2) QKV projection: qkv[M, 3D] = x[M,D] * w_qkv[3D,D]^T  (bf16 out)
    {
        dim3 grid(N3 / 128, M / 128);
        gemm_xwt_kernel<true><<<grid, 256, 0, stream>>>(xb, wqkvb, (void*)qkvb, M, N3, D);
    }

    // 3) Causal flash attention -> context ob[M, D] (bf16)
    {
        dim3 grid(S / 128, 16, B);
        attn_kernel<<<grid, 256, 0, stream>>>(qkvb, ob);
    }

    // 4) Output projection: out[M,D] = ob[M,D] * w_out[D,D]^T  (fp32 out)
    {
        dim3 grid(D / 128, M / 128);
        gemm_xwt_kernel<false><<<grid, 256, 0, stream>>>(ob, woutb, (void*)out, M, D, D);
    }
}
